// Head_86912958201920
// MI455X (gfx1250) — matrix operand order repaired and checked by reading, not compiled, not run on hardware
//
#include <hip/hip_runtime.h>
#include <hip/hip_bf16.h>
#include <math.h>

// Problem constants (fixed by the reference).
constexpr int Tt = 128;   // sequence length
constexpr int Cc = 256;   // embed dim
constexpr int Hh = 64;    // head dim

typedef __attribute__((ext_vector_type(16))) __bf16 v16bf;
typedef __attribute__((ext_vector_type(8)))  __bf16 v8bf;
typedef __attribute__((ext_vector_type(8)))  float  v8f;

// LDS layout. All row strides are multiples of 8 bf16 (16 bytes) so every
// vector access below is a ds_load_b128, and the +8 padding staggers rows
// across banks (stride mod 64 banks = 4 -> 16 lanes tile all 64 banks).
struct __align__(16) Lds {
  union __align__(16) {
    __bf16 wt[3][Hh][Cc + 8];      // W^T for q,k,v (phase 1 only)   101376 B
    __bf16 ps[8][16][Tt + 8];      // per-wave P tiles (phase 2)      34816 B
  } u;
  __bf16 qs [Tt][Hh + 8];          // q (pre-scaled), row-major       18432 B
  __bf16 ks [Tt][Hh + 8];          // k, row-major                    18432 B
  __bf16 vts[Hh][Tt + 8];          // v^T (h-major)                   17408 B
};                                 // total ~152 KB < 320 KB WGP LDS

__device__ __forceinline__ v8f wmma_bf16(v16bf a, v16bf b, v8f c) {
  return __builtin_amdgcn_wmma_f32_16x16x32_bf16(false, a, false, b,
                                                 (short)0, c, false, false);
}

// A-operand (16x32 bf16): per lane two 16B segments at +0 and +16 elems.
__device__ __forceinline__ v16bf ld_a(const __bf16* p) {
  v8bf a = *(const v8bf*)p;
  v8bf b = *(const v8bf*)(p + 16);
  v16bf r;
#pragma unroll
  for (int i = 0; i < 8; ++i) { r[i] = a[i]; r[i + 8] = b[i]; }
  return r;
}

// B-operand (32x16 bf16): per lane two 16B segments at +0 and +16 elems.
__device__ __forceinline__ v16bf ld_b(const __bf16* p) {
  v8bf a = *(const v8bf*)p;
  v8bf b = *(const v8bf*)(p + 16);
  v16bf r;
#pragma unroll
  for (int i = 0; i < 8; ++i) { r[i] = a[i]; r[i + 8] = b[i]; }
  return r;
}

__device__ __forceinline__ void cvt4(v16bf& A, int base, float4 f) {
  A[base + 0] = (__bf16)f.x; A[base + 1] = (__bf16)f.y;
  A[base + 2] = (__bf16)f.z; A[base + 3] = (__bf16)f.w;
}

__global__ __launch_bounds__(256) void head_attn_kernel(
    const float* __restrict__ x,  const float* __restrict__ Wk,
    const float* __restrict__ Wq, const float* __restrict__ Wv,
    float* __restrict__ out) {
  __shared__ Lds lds;

  const int b    = blockIdx.x;
  const int tid  = threadIdx.x;
  const int w    = tid >> 5;          // wave 0..7, owns rows [16w, 16w+16)
  const int lane = tid & 31;
  const int lo   = lane & 15;
  const int hi   = lane >> 4;         // 0: lanes 0-15, 1: lanes 16-31
  const int R    = w * 16;

  // ---- Phase 0: weights -> LDS, transposed, bf16 ------------------------
  for (int i = tid; i < Cc * Hh; i += 256) {
    int c = i >> 6, h = i & 63;       // W is [C][H] row-major
    lds.u.wt[0][h][c] = (__bf16)Wq[i];
    lds.u.wt[1][h][c] = (__bf16)Wk[i];
    lds.u.wt[2][h][c] = (__bf16)Wv[i];
  }
  __syncthreads();

  // ---- Phase 1: q,k,v = x @ W (each x element read from global once) ----
  v8f aq[4] = {}, ak[4] = {}, av[4] = {};
  const float* xrow = x + ((size_t)b * Tt + (R + lo)) * Cc;
#pragma unroll
  for (int kc = 0; kc < 8; ++kc) {    // K = C = 256 in chunks of 32
    const float* p = xrow + kc * 32 + hi * 8;
    float4 f0 = *(const float4*)(p);
    float4 f1 = *(const float4*)(p + 4);
    float4 f2 = *(const float4*)(p + 16);
    float4 f3 = *(const float4*)(p + 20);
    v16bf A;
    cvt4(A, 0, f0); cvt4(A, 4, f1); cvt4(A, 8, f2); cvt4(A, 12, f3);
    const int co = kc * 32 + hi * 8;
#pragma unroll
    for (int nt = 0; nt < 4; ++nt) {  // N = H = 64 in tiles of 16
      aq[nt] = wmma_bf16(A, ld_b(&lds.u.wt[0][nt * 16 + lo][co]), aq[nt]);
      ak[nt] = wmma_bf16(A, ld_b(&lds.u.wt[1][nt * 16 + lo][co]), ak[nt]);
      av[nt] = wmma_bf16(A, ld_b(&lds.u.wt[2][nt * 16 + lo][co]), av[nt]);
    }
  }
  // D layout: element r -> row R + r + 8*hi, col nt*16 + lo.
  const float qscale = 0.125f;        // H^-0.5 = 1/8 (fold into q)
#pragma unroll
  for (int nt = 0; nt < 4; ++nt)
#pragma unroll
    for (int r = 0; r < 8; ++r) {
      int row = R + r + 8 * hi, col = nt * 16 + lo;
      lds.qs[row][col]  = (__bf16)(aq[nt][r] * qscale);
      lds.ks[row][col]  = (__bf16)ak[nt][r];
      lds.vts[col][row] = (__bf16)av[nt][r];   // store v transposed
    }
  __syncthreads();                    // also retires wt before ps aliases it

  // ---- Phase 2a: S = q @ k^T, causal tiles only (nt <= w) ---------------
  v16bf Aq[2];
#pragma unroll
  for (int hc = 0; hc < 2; ++hc)
    Aq[hc] = ld_a(&lds.qs[R + lo][hc * 32 + hi * 8]);

  v8f s[8];
#pragma unroll
  for (int nt = 0; nt < 8; ++nt) {
    v8f acc = {};
    if (nt <= w) {                    // wave-uniform branch (EXEC stays full)
      acc = wmma_bf16(Aq[0], ld_b(&lds.ks[nt * 16 + lo][hi * 8]), acc);
      acc = wmma_bf16(Aq[1], ld_b(&lds.ks[nt * 16 + lo][32 + hi * 8]), acc);
    }
    s[nt] = acc;
  }

  // ---- Phase 2b: causal mask + softmax, fully in-register ---------------
  float mx[8], sm[8];
#pragma unroll
  for (int r = 0; r < 8; ++r) mx[r] = -INFINITY;
#pragma unroll
  for (int nt = 0; nt < 8; ++nt)
#pragma unroll
    for (int r = 0; r < 8; ++r) {
      int row = R + r + 8 * hi, col = nt * 16 + lo;
      float v = (col > row || nt > w) ? -INFINITY : s[nt][r];
      s[nt][r] = v;
      mx[r] = fmaxf(mx[r], v);
    }
#pragma unroll
  for (int r = 0; r < 8; ++r) {       // row reduce within 16-lane half
    mx[r] = fmaxf(mx[r], __shfl_xor(mx[r], 1));
    mx[r] = fmaxf(mx[r], __shfl_xor(mx[r], 2));
    mx[r] = fmaxf(mx[r], __shfl_xor(mx[r], 4));
    mx[r] = fmaxf(mx[r], __shfl_xor(mx[r], 8));
    sm[r] = 0.0f;
  }
#pragma unroll
  for (int nt = 0; nt < 8; ++nt)
#pragma unroll
    for (int r = 0; r < 8; ++r) {
      float e = __expf(s[nt][r] - mx[r]);   // exp(-inf)=0 handles mask
      s[nt][r] = e;
      sm[r] += e;
    }
#pragma unroll
  for (int r = 0; r < 8; ++r) {
    sm[r] += __shfl_xor(sm[r], 1);
    sm[r] += __shfl_xor(sm[r], 2);
    sm[r] += __shfl_xor(sm[r], 4);
    sm[r] += __shfl_xor(sm[r], 8);
    sm[r] = 1.0f / sm[r];             // diagonal always valid -> finite
  }
  // P -> bf16 -> per-wave LDS (D-layout -> A-layout transpose).
#pragma unroll
  for (int nt = 0; nt < 8; ++nt)
#pragma unroll
    for (int r = 0; r < 8; ++r)
      lds.u.ps[w][r + 8 * hi][nt * 16 + lo] = (__bf16)(s[nt][r] * sm[r]);

  // ---- Phase 2c: out = P @ v -------------------------------------------
  v8f o[4] = {};
#pragma unroll
  for (int sc = 0; sc < 4; ++sc) {    // K = T = 128 in chunks of 32
    if (sc * 32 <= R + 15) {          // chunks past the diagonal are all-zero P
      v16bf Ap = ld_a(&lds.u.ps[w][lo][sc * 32 + hi * 8]);
#pragma unroll
      for (int nt = 0; nt < 4; ++nt)
        o[nt] = wmma_bf16(Ap, ld_b(&lds.vts[nt * 16 + lo][sc * 32 + hi * 8]),
                          o[nt]);
    }
  }
#pragma unroll
  for (int nt = 0; nt < 4; ++nt)
#pragma unroll
    for (int r = 0; r < 8; ++r) {
      size_t row = (size_t)b * Tt + R + r + 8 * hi;
      out[row * Hh + nt * 16 + lo] = o[nt][r];
    }
}

extern "C" void kernel_launch(void* const* d_in, const int* in_sizes, int n_in,
                              void* d_out, int out_size, void* d_ws, size_t ws_size,
                              hipStream_t stream) {
  (void)n_in; (void)out_size; (void)d_ws; (void)ws_size;
  const float* x  = (const float*)d_in[0];
  const float* Wk = (const float*)d_in[1];
  const float* Wq = (const float*)d_in[2];
  const float* Wv = (const float*)d_in[3];
  float* out = (float*)d_out;
  const int nb = in_sizes[0] / (Tt * Cc);   // = B = 2048
  head_attn_kernel<<<dim3(nb), dim3(256), 0, stream>>>(x, Wk, Wq, Wv, out);
}
